// LogicalGNNLayer_34514357190805
// MI455X (gfx1250) — matrix operand order repaired and checked by reading, not compiled, not run on hardware
//
#include <hip/hip_runtime.h>

typedef __attribute__((ext_vector_type(16))) _Float16 v16h;
typedef __attribute__((ext_vector_type(8)))  float    v8f;
typedef __attribute__((ext_vector_type(4)))  unsigned v4u;
typedef __attribute__((ext_vector_type(8)))  int      v8i;
typedef __attribute__((ext_vector_type(4)))  int      v4i;

namespace {
constexpr int kT = 16, kB = 4096, kD = 256, kH = 512, kE = 32;
constexpr int kBD = kB * kD;      // 1048576
constexpr int kM  = kT * kB;      // 65536 rows
constexpr float kEPS = 0.1f;
constexpr int kMT = 32;           // M-tile rows per block (2 x 16 sub-tiles)
constexpr int kAP = kD + 8;       // padded agg-tile LDS row stride (halfs) = 264
constexpr int kHP = kH + 8;       // padded hidden-tile LDS row stride (halfs) = 520
}

union Frag {
  v16h v;
  uint4 q[2];
  unsigned u[8];
};

// ---------------- Kernel 1: edge aggregation -> agg (f16) ----------------
__global__ __launch_bounds__(256)
void agg_kernel(const float* __restrict__ term,
                const float* __restrict__ pred,
                const float* __restrict__ ipred,
                const float* __restrict__ signs,
                const int*   __restrict__ heads,
                const int*   __restrict__ tails,
                _Float16*    __restrict__ agg_h) {
  int idx = blockIdx.x * blockDim.x + threadIdx.x;   // over T*B*D, exact grid
  int t = idx >> 20;            // idx / (B*D)
  int r = idx & (kBD - 1);      // idx % (B*D)
  float acc = kEPS * term[idx];
#pragma unroll 4
  for (int e = 0; e < kE; ++e) {
    int he = heads[e];
    int te = tails[e];
    float s = signs[e];
    if (te == t) acc += s * (term[he * kBD + r] + pred[e * kBD + r]);
    if (he == t) acc += s * (term[te * kBD + r] + ipred[e * kBD + r]);
  }
  agg_h[idx] = (_Float16)acc;
}

// -------- Kernel 0: cast+transpose weights to f16 [N][K] layouts ---------
__global__ __launch_bounds__(256)
void prep_kernel(const float* __restrict__ W1, const float* __restrict__ W2,
                 _Float16* __restrict__ W1T, _Float16* __restrict__ W2T) {
  int idx = blockIdx.x * blockDim.x + threadIdx.x;   // < D*H == H*D == 131072
  {
    int k = idx / kH, n = idx % kH;                  // W1 [D][H] -> W1T [H][D]
    W1T[n * kD + k] = (_Float16)W1[idx];
  }
  {
    int k = idx / kD, n = idx % kD;                  // W2 [H][D] -> W2T [D][H]
    W2T[n * kH + k] = (_Float16)W2[idx];
  }
}

// ---------------- Kernel 2: fused MLP with f16 WMMA + TDM ----------------
__global__ __launch_bounds__(256)
void mlp_kernel(const _Float16* __restrict__ agg_h,
                const _Float16* __restrict__ W1T,
                const float*    __restrict__ bias1,
                const _Float16* __restrict__ W2T,
                const float*    __restrict__ bias2,
                float*          __restrict__ out) {
  __shared__ _Float16 aggtile[kMT * kAP];            // 32 x 264 halfs = 16.5 KB
  __shared__ _Float16 hid[kMT * kHP];                // 32 x 520 halfs = 32.5 KB

  const int tid  = threadIdx.x;
  const int wave = tid >> 5;
  const int lane = tid & 31;
  const int ln   = lane & 15;                        // N col / M row within half-wave
  const int g    = lane >> 4;                        // lane-group
  const int Mbase = blockIdx.x * kMT;

  // ---- Stage agg tile (rows Mbase..Mbase+31, K=0..255) into LDS ----------
#if __has_builtin(__builtin_amdgcn_tensor_load_to_lds) && __has_builtin(__builtin_amdgcn_s_wait_tensorcnt)
  if (tid == 0) {
    // Tensor DMA descriptor (D#): 2D tile of 2-byte elements, 256 x 32,
    // row stride 256 elems; LDS padding: every 128 DWORDs (one row = 512 B)
    // insert 4 DWORDs (8 halfs) -> LDS row stride = 264 halfs = kAP.
    unsigned ldsOff = (unsigned)(size_t)(&aggtile[0]);
    unsigned long long ga =
        (unsigned long long)(size_t)(agg_h + (size_t)Mbase * kD);
    v4u g0;
    g0.x = 1u;                                        // count = 1 valid descriptor
    g0.y = ldsOff;                                    // lds_addr
    g0.z = (unsigned)(ga & 0xFFFFFFFFull);            // global_addr[31:0]
    g0.w = (unsigned)((ga >> 32) & 0x1FFFFFFull)      // global_addr[56:32]
         | (2u << 30);                                // type = 2 ("image")
    v8i g1;
    g1[0] = (int)((1u << 16)        // data_size = 1 -> 2 bytes
                | (1u << 20)        // pad_enable
                | (6u << 22)        // pad_interval: 2^(6+1)=128 DWORDs
                | (3u << 25));      // pad_amount: 4 DWORDs
    g1[1] = (int)(256u << 16);      // tensor_dim0[15:0] = 256
    g1[2] = (int)(32u  << 16);      // tensor_dim0 hi = 0 | tensor_dim1 = 32
    g1[3] = (int)(256u << 16);      // tensor_dim1 hi = 0 | tile_dim0 = 256
    g1[4] = 32;                     // tile_dim1 = 32, tile_dim2 = 0
    g1[5] = 256;                    // tensor_dim0_stride = 256
    g1[6] = 0;
    g1[7] = 0;
    v4i gz; gz[0] = 0; gz[1] = 0; gz[2] = 0; gz[3] = 0;
    v8i gz8; gz8[0] = 0; gz8[1] = 0; gz8[2] = 0; gz8[3] = 0;
    gz8[4] = 0; gz8[5] = 0; gz8[6] = 0; gz8[7] = 0;
    __builtin_amdgcn_tensor_load_to_lds(g0, g1, gz, gz, gz8, 0);
    __builtin_amdgcn_s_wait_tensorcnt(0);
  }
#else
  {
    const unsigned* src = (const unsigned*)(agg_h + (size_t)Mbase * kD);
    unsigned* dst = (unsigned*)aggtile;
#pragma unroll
    for (int it = 0; it < (kMT * kD / 2) / 256; ++it) {
      int i = it * 256 + tid;                        // uint index over 32x128
      int row = i >> 7, kp = i & 127;
      dst[(row * kAP) / 2 + kp] = src[row * (kD / 2) + kp];
    }
  }
#endif
  __syncthreads();

  // ---- GEMM1: hid = relu(agg @ W1 + b1); wave owns 4 of 32 N-tiles,
  //      each B fragment reused across 2 M sub-tiles ----------------------
#pragma unroll
  for (int j = 0; j < 4; ++j) {
    const int n0  = (wave * 4 + j) * 16;
    const int col = n0 + ln;
    const _Float16* bcol = W1T + col * kD;           // B: lane=col, contiguous K
    v8f c0 = {}, c1 = {};
#pragma unroll
    for (int kt = 0; kt < 8; ++kt) {
      Frag b;
      b.q[0] = *(const uint4*)(bcol + kt * 32 + 8 * g);
      b.q[1] = *(const uint4*)(bcol + kt * 32 + 16 + 8 * g);
      Frag a0, a1;
      const _Float16* ar0 = aggtile + ln * kAP;            // sub-tile m=0
      const _Float16* ar1 = aggtile + (16 + ln) * kAP;     // sub-tile m=1
      a0.q[0] = *(const uint4*)(ar0 + kt * 32 + 8 * g);
      a0.q[1] = *(const uint4*)(ar0 + kt * 32 + 16 + 8 * g);
      a1.q[0] = *(const uint4*)(ar1 + kt * 32 + 8 * g);
      a1.q[1] = *(const uint4*)(ar1 + kt * 32 + 16 + 8 * g);
      c0 = __builtin_amdgcn_wmma_f32_16x16x32_f16(false, a0.v, false, b.v,
                                                  (short)0, c0, false, false);
      c1 = __builtin_amdgcn_wmma_f32_16x16x32_f16(false, a1.v, false, b.v,
                                                  (short)0, c1, false, false);
    }
    const float bv = bias1[col];
#pragma unroll
    for (int v = 0; v < 8; ++v) {                    // C/D: VGPR v -> row v+8g
      float h0 = c0[v] + bv;
      float h1 = c1[v] + bv;
      h0 = h0 > 0.f ? h0 : 0.f;
      h1 = h1 > 0.f ? h1 : 0.f;
      hid[(v + 8 * g) * kHP + col]        = (_Float16)h0;
      hid[(16 + v + 8 * g) * kHP + col]   = (_Float16)h1;
    }
  }
  __syncthreads();

  // ---- GEMM2: out = hid @ W2 + b2; wave owns 2 of 16 N-tiles -------------
#pragma unroll
  for (int j = 0; j < 2; ++j) {
    const int n0  = (wave * 2 + j) * 16;
    const int col = n0 + ln;
    const _Float16* bcol = W2T + col * kH;
    v8f c0 = {}, c1 = {};
#pragma unroll 4
    for (int kt = 0; kt < 16; ++kt) {
      Frag b;
      b.q[0] = *(const uint4*)(bcol + kt * 32 + 8 * g);
      b.q[1] = *(const uint4*)(bcol + kt * 32 + 16 + 8 * g);
      Frag a0, a1;
      const _Float16* ar0 = hid + ln * kHP;
      const _Float16* ar1 = hid + (16 + ln) * kHP;
      a0.q[0] = *(const uint4*)(ar0 + kt * 32 + 8 * g);
      a0.q[1] = *(const uint4*)(ar0 + kt * 32 + 16 + 8 * g);
      a1.q[0] = *(const uint4*)(ar1 + kt * 32 + 8 * g);
      a1.q[1] = *(const uint4*)(ar1 + kt * 32 + 16 + 8 * g);
      c0 = __builtin_amdgcn_wmma_f32_16x16x32_f16(false, a0.v, false, b.v,
                                                  (short)0, c0, false, false);
      c1 = __builtin_amdgcn_wmma_f32_16x16x32_f16(false, a1.v, false, b.v,
                                                  (short)0, c1, false, false);
    }
    const float bv = bias2[col];
#pragma unroll
    for (int v = 0; v < 8; ++v) {
      out[(Mbase + v + 8 * g) * kD + col]      = c0[v] + bv;
      out[(Mbase + 16 + v + 8 * g) * kD + col] = c1[v] + bv;
    }
  }
}

extern "C" void kernel_launch(void* const* d_in, const int* in_sizes, int n_in,
                              void* d_out, int out_size, void* d_ws, size_t ws_size,
                              hipStream_t stream) {
  const float* term  = (const float*)d_in[0];
  const float* pred  = (const float*)d_in[1];
  const float* ipred = (const float*)d_in[2];
  const float* signs = (const float*)d_in[3];
  const float* W1    = (const float*)d_in[4];
  const float* b1    = (const float*)d_in[5];
  const float* W2    = (const float*)d_in[6];
  const float* b2    = (const float*)d_in[7];
  const int*   heads = (const int*)d_in[8];
  const int*   tails = (const int*)d_in[9];
  float* out = (float*)d_out;

  // workspace: agg_h (32 MB f16) | W1T (256 KB f16) | W2T (256 KB f16)
  char* ws = (char*)d_ws;
  _Float16* agg_h = (_Float16*)ws;
  _Float16* W1T   = (_Float16*)(ws + (size_t)kM * kD * sizeof(_Float16));
  _Float16* W2T   = W1T + kD * kH;

  prep_kernel<<<(kD * kH) / 256, 256, 0, stream>>>(W1, W2, W1T, W2T);
  agg_kernel<<<(kM * kD) / 256, 256, 0, stream>>>(term, pred, ipred, signs,
                                                  heads, tails, agg_h);
  mlp_kernel<<<kM / kMT, 256, 0, stream>>>(agg_h, W1T, b1, W2T, b2, out);
}
